// BiLSTM_CRF_86388972192061
// MI455X (gfx1250) — matrix operand order repaired and checked by reading, not compiled, not run on hardware
//
#include <hip/hip_runtime.h>
#include <hip/hip_bf16.h>

// ---------------------------------------------------------------------------
// BiLSTM-CRF for MI455X (gfx1250, wave32, WMMA 16x16x32 bf16 + TDM prefetch)
// ---------------------------------------------------------------------------
#define S_LEN 512
#define B_SZ  32
#define E_DIM 256
#define H_DIM 256
#define G_DIM 1024   // 4*H
#define T_TAG 16
#define ROWS  (S_LEN * B_SZ)   // 16384

typedef __attribute__((ext_vector_type(16))) __bf16 bf16x16;
typedef __attribute__((ext_vector_type(8)))  float  f32x8;

#if defined(__HIP_DEVICE_COMPILE__) && __has_builtin(__builtin_amdgcn_tensor_load_to_lds) && __has_builtin(__builtin_amdgcn_s_wait_tensorcnt)
#define HAVE_TDM 1
#else
#define HAVE_TDM 0
#endif

union AFrag {
    bf16x16 bf;
    uint4   q[2];
};

// A-fragment (16x32 bf16, row-major source, K contiguous):
// lanes 0-15 : rows 0-15, K = kbase+0..15
// lanes 16-31: rows 0-15, K = kbase+16..31
__device__ __forceinline__ bf16x16 load_a_frag(const __bf16* base, int row_stride,
                                               int lane, int kbase) {
    const int row = lane & 15;
    const int ko  = kbase + ((lane < 16) ? 0 : 16);
    return *(const bf16x16*)(base + row * row_stride + ko);
}

// B-fragment (32x16 bf16) from a ROW-MAJOR weight W[n][k] (k contiguous),
// i.e. B[k][n] = W[n][k]: lane holds column n = lane%16; lanes 0-15 K=kbase..+15,
// lanes 16-31 K=kbase+16..+31; 16 consecutive K -> one 32-byte contiguous load.
__device__ __forceinline__ bf16x16 load_b_frag(const __bf16* base, int row_stride,
                                               int lane, int kbase) {
    const int n  = lane & 15;
    const int ko = kbase + ((lane < 16) ? 0 : 16);
    return *(const bf16x16*)(base + n * row_stride + ko);
}

__device__ __forceinline__ float sigmoidf_fast(float x) {
    return 1.0f / (1.0f + __expf(-x));
}

// --- CDNA5 split workgroup barrier + counter waits (inline asm) -------------
__device__ __forceinline__ void wg_barrier_signal() {
    asm volatile("s_barrier_signal -1" ::: "memory");
}
__device__ __forceinline__ void wg_barrier_wait() {
    asm volatile("s_barrier_wait -1" ::: "memory");
}
__device__ __forceinline__ void wait_dscnt0() {
    asm volatile("s_wait_dscnt 0x0" ::: "memory");
}

#if HAVE_TDM
typedef __attribute__((ext_vector_type(4))) unsigned int v4u;
typedef __attribute__((ext_vector_type(8))) int          v8i;
typedef __attribute__((ext_vector_type(4))) int          v4i;

// 1-D TDM transfer: 16384 f32 (64 KB), contiguous global -> contiguous LDS.
__device__ __forceinline__ void tdm_load_pre_tile(const float* gsrc, float* ldst) {
    const unsigned long long ga = (unsigned long long)(uintptr_t)gsrc;
    const unsigned int       la = (unsigned int)(uintptr_t)ldst;  // LDS byte offset
    v4u g0;
    g0[0] = 1u;                                   // count=1 (valid user descriptor)
    g0[1] = la;                                   // lds_addr
    g0[2] = (unsigned int)ga;                     // global_addr[31:0]
    g0[3] = (unsigned int)((ga >> 32) & 0x01FFFFFFu) | 0x80000000u; // [56:32] | type=2
    v8i g1;
    g1[0] = (int)(2u << 16);                      // data_size = 4 bytes
    g1[1] = (int)(16384u << 16);                  // tensor_dim0[15:0] << 16
    g1[2] = (int)(1u << 16);                      // tensor_dim0[31:16]=0 | tensor_dim1=1
    g1[3] = (int)(16384u << 16);                  // tile_dim0 = 16384
    g1[4] = 0;                                    // tile_dim1 = tile_dim2 = 0 (1-D)
    g1[5] = 16384;                                // tensor_dim0_stride
    g1[6] = 0;
    g1[7] = 0;
    v4i z = {0, 0, 0, 0};
#if defined(__clang_major__) && (__clang_major__ >= 23)
    v8i z8 = {0, 0, 0, 0, 0, 0, 0, 0};
    __builtin_amdgcn_tensor_load_to_lds(g0, g1, z, z, z8, 0);
#else
    __builtin_amdgcn_tensor_load_to_lds(g0, g1, z, z, 0);
#endif
}
#endif

// ---------------------------------------------------------------------------
// Kernel 0: convert weights to bf16, fold biases (bi+bh)
// ---------------------------------------------------------------------------
__global__ void k_convert(const float* __restrict__ Wi_f, const float* __restrict__ Wh_f,
                          const float* __restrict__ Wi_b, const float* __restrict__ Wh_b,
                          const float* __restrict__ Wt,
                          const float* __restrict__ bi_f, const float* __restrict__ bh_f,
                          const float* __restrict__ bi_b, const float* __restrict__ bh_b,
                          __bf16* __restrict__ Wi_bf, __bf16* __restrict__ Wh_bf,
                          __bf16* __restrict__ Wt_bf, float* __restrict__ bias) {
    const int i = blockIdx.x * blockDim.x + threadIdx.x;
    const int WN = G_DIM * E_DIM;  // 262144
    if (i < WN) {
        Wi_bf[i]      = (__bf16)Wi_f[i];
        Wi_bf[WN + i] = (__bf16)Wi_b[i];
        Wh_bf[i]      = (__bf16)Wh_f[i];
        Wh_bf[WN + i] = (__bf16)Wh_b[i];
    }
    if (i < T_TAG * (2 * H_DIM)) Wt_bf[i] = (__bf16)Wt[i];
    if (i < G_DIM) {
        bias[i]         = bi_f[i] + bh_f[i];
        bias[G_DIM + i] = bi_b[i] + bh_b[i];
    }
}

// ---------------------------------------------------------------------------
// Kernel 1: embedding gather + f32->bf16 ; xbf[s*32+b][e]
// ---------------------------------------------------------------------------
__global__ void k_gather(const int* __restrict__ tokens, const float* __restrict__ embed,
                         __bf16* __restrict__ xbf) {
    const int row = blockIdx.x;            // 0..16383  (s*32 + b)
    const int s = row >> 5;
    const int b = row & 31;
    const int tok = tokens[b * S_LEN + s];
    const int e = threadIdx.x;             // 0..255
    xbf[(size_t)row * E_DIM + e] = (__bf16)embed[(size_t)tok * E_DIM + e];
}

// ---------------------------------------------------------------------------
// Kernel 2: input GEMM  pre[d][row][g] = xbf[row][:] . Wi[d][g][:] + bias[d][g]
// grid (1024 Mtiles, 2 dirs), 256 threads = 8 waves; wave: 8 N-tiles as 4 pairs
// (two independent WMMA chains per pair for ILP / load overlap).
// ---------------------------------------------------------------------------
__global__ __launch_bounds__(256) void k_pregemm(const __bf16* __restrict__ xbf,
                                                 const __bf16* __restrict__ Wi_bf,
                                                 const float* __restrict__ bias,
                                                 float* __restrict__ pre) {
    const int mt   = blockIdx.x;          // 0..1023
    const int dir  = blockIdx.y;          // 0..1
    const int wave = threadIdx.x >> 5;    // 0..7
    const int lane = threadIdx.x & 31;
    const int m_hi = (lane < 16) ? 0 : 8;
    const int nl   = lane & 15;

    bf16x16 afr[8];
    const __bf16* arow = xbf + (size_t)mt * 16 * E_DIM;
#pragma unroll
    for (int kt = 0; kt < 8; ++kt) afr[kt] = load_a_frag(arow, E_DIM, lane, kt * 32);

    const __bf16* Wd = Wi_bf + (size_t)dir * (G_DIM * E_DIM);
    const float*  bd = bias + dir * G_DIM;
    float*        pd = pre + (size_t)dir * ((size_t)ROWS * G_DIM);

    for (int np = 0; np < 4; ++np) {
        const int nt0 = wave * 8 + np * 2;
        const int nt1 = nt0 + 1;
        const float bv0 = bd[nt0 * 16 + nl];
        const float bv1 = bd[nt1 * 16 + nl];
        f32x8 acc0, acc1;
#pragma unroll
        for (int r = 0; r < 8; ++r) { acc0[r] = bv0; acc1[r] = bv1; }
        const __bf16* w0 = Wd + (size_t)nt0 * 16 * E_DIM;
        const __bf16* w1 = Wd + (size_t)nt1 * 16 * E_DIM;
#pragma unroll
        for (int kt = 0; kt < 8; ++kt) {
            bf16x16 b0 = load_b_frag(w0, E_DIM, lane, kt * 32);
            bf16x16 b1 = load_b_frag(w1, E_DIM, lane, kt * 32);
            acc0 = __builtin_amdgcn_wmma_f32_16x16x32_bf16(false, afr[kt], false, b0,
                                                           (short)0, acc0, false, false);
            acc1 = __builtin_amdgcn_wmma_f32_16x16x32_bf16(false, afr[kt], false, b1,
                                                           (short)0, acc1, false, false);
        }
        float* out0 = pd + ((size_t)mt * 16 + m_hi) * G_DIM + nt0 * 16 + nl;
        float* out1 = pd + ((size_t)mt * 16 + m_hi) * G_DIM + nt1 * 16 + nl;
#pragma unroll
        for (int r = 0; r < 8; ++r) { out0[(size_t)r * G_DIM] = acc0[r];
                                      out1[(size_t)r * G_DIM] = acc1[r]; }
    }
}

// ---------------------------------------------------------------------------
// Kernel 3: recurrent LSTM (both directions).
// grid = 4 blocks: blockIdx.x = dir*2 + mtile (16 batch rows each), 16 waves.
// Wave w owns all 4 gate tiles of h-chunk w -> gate fusion is wave-local.
// pre tiles (HBM) are TDM-prefetched 2 steps ahead into a triple-buffered LDS
// staging area on TENSORcnt (decoupled from the L2 B-fragment LOADcnt waits).
// ---------------------------------------------------------------------------
__global__ __launch_bounds__(512) void k_lstm(const float* __restrict__ pre,
                                              const __bf16* __restrict__ Wh_bf,
                                              __bf16* __restrict__ feats) {
    const int dir   = blockIdx.x >> 1;
    const int mtile = blockIdx.x & 1;
    const int wave  = threadIdx.x >> 5;   // 0..15 = h-chunk
    const int lane  = threadIdx.x & 31;
    const int m_hi  = (lane < 16) ? 0 : 8;
    const int nl    = lane & 15;
    const int hunit = wave * 16 + nl;     // 0..255

    __shared__ __align__(32) __bf16 h_lds[16 * H_DIM];          // 8 KB
#if HAVE_TDM
    __shared__ __align__(128) float pre_lds[3][16 * G_DIM];     // 3 x 64 KB
#endif

    for (int i = threadIdx.x; i < 16 * H_DIM / 2; i += 512)
        ((unsigned int*)h_lds)[i] = 0u;

    const float*  pre_d = pre + (size_t)dir * ((size_t)ROWS * G_DIM);
    const __bf16* Wh_d  = Wh_bf + (size_t)dir * (G_DIM * H_DIM);

#if HAVE_TDM
    if (threadIdx.x < 32) {   // wave 0 drives the TDM
        const int s0 = (dir == 0) ? 0 : (S_LEN - 1);
        const int s1 = (dir == 0) ? 1 : (S_LEN - 2);
        tdm_load_pre_tile(pre_d + ((size_t)s0 * B_SZ + mtile * 16) * G_DIM, &pre_lds[0][0]);
        tdm_load_pre_tile(pre_d + ((size_t)s1 * B_SZ + mtile * 16) * G_DIM, &pre_lds[1][0]);
        __builtin_amdgcn_s_wait_tensorcnt(1);   // step-0 tile resident
    }
#endif
    __syncthreads();

    float c[8];
#pragma unroll
    for (int r = 0; r < 8; ++r) c[r] = 0.0f;

    for (int step = 0; step < S_LEN; ++step) {
        const int s = (dir == 0) ? step : (S_LEN - 1 - step);

        // ---- z init from pre (LDS staging buffer or global fallback) ----
        f32x8 acc[4];
#if HAVE_TDM
        const float* pl = &pre_lds[step % 3][0] + (size_t)m_hi * G_DIM;
#pragma unroll
        for (int g = 0; g < 4; ++g) {
            const float* pg = pl + g * H_DIM + hunit;
#pragma unroll
            for (int r = 0; r < 8; ++r) acc[g][r] = pg[(size_t)r * G_DIM];
        }
        if (threadIdx.x < 32 && step + 2 < S_LEN) {  // prefetch step+2
            const int sn = (dir == 0) ? (step + 2) : (S_LEN - 3 - step);
            tdm_load_pre_tile(pre_d + ((size_t)sn * B_SZ + mtile * 16) * G_DIM,
                              &pre_lds[(step + 2) % 3][0]);
        }
#else
        const float* prow = pre_d + ((size_t)s * B_SZ + mtile * 16 + m_hi) * G_DIM;
#pragma unroll
        for (int g = 0; g < 4; ++g) {
            const float* pg = prow + g * H_DIM + hunit;
#pragma unroll
            for (int r = 0; r < 8; ++r) acc[g][r] = pg[(size_t)r * G_DIM];
        }
#endif

        // ---- z += h . Wh^T  (K=256, double-buffered B fragments) ----
        bf16x16 bcur[4];
#pragma unroll
        for (int g = 0; g < 4; ++g)
            bcur[g] = load_b_frag(Wh_d + (size_t)(g * H_DIM + wave * 16) * H_DIM,
                                  H_DIM, lane, 0);
#pragma unroll
        for (int kt = 0; kt < 8; ++kt) {
            bf16x16 a = load_a_frag(h_lds, H_DIM, lane, kt * 32);
            bf16x16 bnxt[4];
            if (kt < 7) {
#pragma unroll
                for (int g = 0; g < 4; ++g)
                    bnxt[g] = load_b_frag(Wh_d + (size_t)(g * H_DIM + wave * 16) * H_DIM,
                                          H_DIM, lane, (kt + 1) * 32);
            }
#pragma unroll
            for (int g = 0; g < 4; ++g)
                acc[g] = __builtin_amdgcn_wmma_f32_16x16x32_bf16(false, a, false, bcur[g],
                                                                 (short)0, acc[g],
                                                                 false, false);
            if (kt < 7) {
#pragma unroll
                for (int g = 0; g < 4; ++g) bcur[g] = bnxt[g];
            }
        }

        // ---- split barrier #1: my LDS reads done; gate math overlaps ----
        wait_dscnt0();
        wg_barrier_signal();

        float hval[8];
#pragma unroll
        for (int r = 0; r < 8; ++r) {
            const float zi = acc[0][r], zf = acc[1][r], zg = acc[2][r], zo = acc[3][r];
            const float cc = sigmoidf_fast(zf) * c[r] + sigmoidf_fast(zi) * tanhf(zg);
            c[r] = cc;
            hval[r] = sigmoidf_fast(zo) * tanhf(cc);
        }

        wg_barrier_wait();     // all waves finished reading h_lds

        // ---- publish h; split barrier #2 overlaps global feats stores ----
#pragma unroll
        for (int r = 0; r < 8; ++r)
            h_lds[(m_hi + r) * H_DIM + hunit] = (__bf16)hval[r];
        wait_dscnt0();
#if HAVE_TDM
        if (threadIdx.x < 32) {   // step+1 staging tile must be resident
            if (step + 2 < S_LEN) __builtin_amdgcn_s_wait_tensorcnt(1);
            else                  __builtin_amdgcn_s_wait_tensorcnt(0);
        }
#endif
        wg_barrier_signal();
#pragma unroll
        for (int r = 0; r < 8; ++r)
            feats[(((size_t)s * B_SZ) + mtile * 16 + m_hi + r) * (2 * H_DIM)
                  + dir * H_DIM + hunit] = (__bf16)hval[r];
        wg_barrier_wait();
    }
}

// ---------------------------------------------------------------------------
// Kernel 4: emissions GEMM  emis[row][t] = feats[row][:] . Wt[t][:] + bt[t]
// M=16384 (1024 tiles), N=1 tile, K=512; two independent K chains per wave.
// ---------------------------------------------------------------------------
__global__ __launch_bounds__(256) void k_emis(const __bf16* __restrict__ feats,
                                              const __bf16* __restrict__ Wt_bf,
                                              const float* __restrict__ bt,
                                              float* __restrict__ emis) {
    const int wave = threadIdx.x >> 5;
    const int lane = threadIdx.x & 31;
    const int mt   = blockIdx.x * 8 + wave;    // 0..1023
    const int m_hi = (lane < 16) ? 0 : 8;
    const int nl   = lane & 15;

    f32x8 acc0, acc1;
    const float bv = bt[nl];
#pragma unroll
    for (int r = 0; r < 8; ++r) { acc0[r] = bv; acc1[r] = 0.0f; }

    const __bf16* arow = feats + (size_t)mt * 16 * (2 * H_DIM);
#pragma unroll
    for (int kt = 0; kt < 16; kt += 2) {
        bf16x16 a0 = load_a_frag(arow, 2 * H_DIM, lane, kt * 32);
        bf16x16 b0 = load_b_frag(Wt_bf, 2 * H_DIM, lane, kt * 32);
        bf16x16 a1 = load_a_frag(arow, 2 * H_DIM, lane, (kt + 1) * 32);
        bf16x16 b1 = load_b_frag(Wt_bf, 2 * H_DIM, lane, (kt + 1) * 32);
        acc0 = __builtin_amdgcn_wmma_f32_16x16x32_bf16(false, a0, false, b0,
                                                       (short)0, acc0, false, false);
        acc1 = __builtin_amdgcn_wmma_f32_16x16x32_bf16(false, a1, false, b1,
                                                       (short)0, acc1, false, false);
    }
    float* out = emis + ((size_t)mt * 16 + m_hi) * T_TAG + nl;
#pragma unroll
    for (int r = 0; r < 8; ++r) out[(size_t)r * T_TAG] = acc0[r] + acc1[r];
}

// ---------------------------------------------------------------------------
// Kernel 5: CRF forward (log-partition). 1 block, 512 threads = (batch, tag).
// ---------------------------------------------------------------------------
__global__ __launch_bounds__(512) void k_crf(const float* __restrict__ emis,
                                             const float* __restrict__ trans,
                                             float* __restrict__ out) {
    const int t = threadIdx.x & 15;
    const int b = threadIdx.x >> 4;        // 0..31
    __shared__ float tr[T_TAG * T_TAG];
    __shared__ float alpha[B_SZ * T_TAG];

    if (threadIdx.x < T_TAG * T_TAG) tr[threadIdx.x] = trans[threadIdx.x];
    alpha[b * T_TAG + t] = (t == 14) ? 0.0f : -10000.0f;  // START = 14
    __syncthreads();

    for (int s = 0; s < S_LEN; ++s) {
        const float emit = emis[((size_t)s * B_SZ + b) * T_TAG + t];
        float v[T_TAG];
        float m = -3.0e38f;
#pragma unroll
        for (int p = 0; p < T_TAG; ++p) {
            v[p] = alpha[b * T_TAG + p] + tr[t * T_TAG + p];
            m = fmaxf(m, v[p]);
        }
        float sum = 0.0f;
#pragma unroll
        for (int p = 0; p < T_TAG; ++p) sum += __expf(v[p] - m);
        const float na = emit + m + __logf(sum);
        __syncthreads();
        alpha[b * T_TAG + t] = na;
        __syncthreads();
    }

    if (t == 0) {  // STOP = 15
        float v[T_TAG];
        float m = -3.0e38f;
#pragma unroll
        for (int p = 0; p < T_TAG; ++p) {
            v[p] = alpha[b * T_TAG + p] + tr[15 * T_TAG + p];
            m = fmaxf(m, v[p]);
        }
        float sum = 0.0f;
#pragma unroll
        for (int p = 0; p < T_TAG; ++p) sum += __expf(v[p] - m);
        out[b] = m + __logf(sum);
    }
}

// ---------------------------------------------------------------------------
extern "C" void kernel_launch(void* const* d_in, const int* in_sizes, int n_in,
                              void* d_out, int out_size, void* d_ws, size_t ws_size,
                              hipStream_t stream) {
    (void)in_sizes; (void)n_in; (void)out_size; (void)ws_size;
    const int*   tokens = (const int*)d_in[0];
    const float* embed  = (const float*)d_in[1];
    const float* Wi_f   = (const float*)d_in[2];
    const float* Wh_f   = (const float*)d_in[3];
    const float* bi_f   = (const float*)d_in[4];
    const float* bh_f   = (const float*)d_in[5];
    const float* Wi_b   = (const float*)d_in[6];
    const float* Wh_b   = (const float*)d_in[7];
    const float* bi_b   = (const float*)d_in[8];
    const float* bh_b   = (const float*)d_in[9];
    const float* Wt     = (const float*)d_in[10];
    const float* bt     = (const float*)d_in[11];
    const float* trans  = (const float*)d_in[12];
    float* out = (float*)d_out;

    // Workspace carve (all sizes 256B-aligned)
    char* p = (char*)d_ws;
    __bf16* xbf   = (__bf16*)p; p += (size_t)ROWS * E_DIM * 2;          // 8 MB
    __bf16* Wi_bf = (__bf16*)p; p += (size_t)2 * G_DIM * E_DIM * 2;     // 1 MB
    __bf16* Wh_bf = (__bf16*)p; p += (size_t)2 * G_DIM * H_DIM * 2;     // 1 MB
    __bf16* Wt_bf = (__bf16*)p; p += (size_t)T_TAG * 2 * H_DIM * 2;     // 16 KB
    float*  bias  = (float*)p;  p += (size_t)2 * G_DIM * 4;             // 8 KB
    float*  pre   = (float*)p;  p += (size_t)2 * ROWS * G_DIM * 4;      // 128 MB
    __bf16* feats = (__bf16*)p; p += (size_t)ROWS * 2 * H_DIM * 2;      // 16 MB
    float*  emis  = (float*)p;  p += (size_t)ROWS * T_TAG * 4;          // 1 MB

    k_convert<<<(G_DIM * E_DIM + 255) / 256, 256, 0, stream>>>(
        Wi_f, Wh_f, Wi_b, Wh_b, Wt, bi_f, bh_f, bi_b, bh_b,
        Wi_bf, Wh_bf, Wt_bf, bias);

    k_gather<<<ROWS, E_DIM, 0, stream>>>(tokens, embed, xbf);

    k_pregemm<<<dim3(ROWS / 16, 2), 256, 0, stream>>>(xbf, Wi_bf, bias, pre);

    k_lstm<<<4, 512, 0, stream>>>(pre, Wh_bf, feats);

    k_emis<<<ROWS / 16 / 8, 256, 0, stream>>>(feats, Wt_bf, bt, emis);

    k_crf<<<1, 512, 0, stream>>>(emis, trans, out);
}
